// TransformerModel_48713519071575
// MI455X (gfx1250) — matrix operand order repaired and checked by reading, not compiled, not run on hardware
//
#include <hip/hip_runtime.h>
#include <hip/hip_bf16.h>

// ---------------------------------------------------------------------------
// Transformer encoder-decoder forward for MI455X (gfx1250, wave32, WMMA).
// All matmuls run through one batched-strided WMMA GEMM kernel using
// V_WMMA_F32_16X16X32_F16 (fp16 inputs, fp32 accumulation).
// B (weight) tiles are staged with CDNA5 async global->LDS copies
// (global_load_async_to_lds_b32, GVS saddr form, ASYNCcnt) into a
// double-buffered raw region so the DMA overlaps a full K-step of WMMA work.
// A tiles use vectorized b128 loads + fp16 conversion with running pointers.
// ---------------------------------------------------------------------------

static constexpr int E_   = 1024;
static constexpr int H_   = 16;
static constexpr int DH_  = 64;
static constexpr int DFF_ = 4096;
static constexpr int LE_  = 4;
static constexpr int LD_  = 4;
static constexpr int B_   = 4;
static constexpr int S_   = 512;          // S_SRC == S_TGT
static constexpr int M_   = B_ * S_;      // 2048 tokens

typedef _Float16     v16h  __attribute__((ext_vector_type(16)));
typedef _Float16     h4    __attribute__((ext_vector_type(4)));
typedef float        f32x8 __attribute__((ext_vector_type(8)));
typedef unsigned int u32x4 __attribute__((ext_vector_type(4)));

union Frag {
  v16h  h;
  u32x4 q[2];
};

// GEMM block tile: 128(M) x BNv(N) x 32(K), 256 threads = 8 wave32 waves
// arranged 4(m) x 2(n); each wave owns 32 x (BNv/2) output elements
// = 2 x (BNv/32) fragments of 16x16 computed by V_WMMA_F32_16X16X32_F16.
#define BM 128
#define BK 32
#define LDA_S 40   // LDS row stride (halves) for A tile  (16B-aligned rows)
#define LDB_S 40   // LDS row stride (halves) for Bt tile (stored transposed)

// Generic batched GEMM: C[z1,z2][m,n] = alpha * sum_k A*B (+bias +residual, relu)
//   A index: z1*saz1 + z2*saz2 + m*lda + k                (k-contiguous)
//   B index: z1*sbz1 + z2*sbz2 + k*ldb + (n>>6)*hsb + (n&63)*bsn
//   C index: z1*scz1 + z2*scz2 + m*ldc + n                (residual likewise)
template <int BNv>
__global__ __launch_bounds__(256) void wmma_gemm(
    const float* __restrict__ A, const float* __restrict__ B,
    float* __restrict__ C, const float* __restrict__ bias,
    const float* __restrict__ resid,
    int M, int N, int K,
    long long saz1, long long saz2, long long lda,
    long long sbz1, long long sbz2, long long ldb, long long hsb, long long bsn,
    long long scz1, long long scz2, long long ldc,
    int nz2, float alpha, int relu)
{
#if defined(__gfx1250__)
  constexpr int FN = BNv / 32;               // fragment columns per wave
  constexpr int SA_OFF    = 0;               // fp16 A tile
  constexpr int SA_BYTES  = BM * LDA_S * 2;  // 10240
  constexpr int SBR_OFF   = SA_OFF + SA_BYTES;     // raw fp32 B tiles (x2)
  constexpr int SBR_BYTES = BK * BNv * 4;
  constexpr int SBT_OFF   = SBR_OFF + 2 * SBR_BYTES;  // fp16 Bt tile
  constexpr int SBT_BYTES = BNv * LDB_S * 2;
  constexpr int SMEM_BYTES = SBT_OFF + SBT_BYTES;

  __shared__ __align__(16) char smem[SMEM_BYTES];
  _Float16* sA  = (_Float16*)&smem[SA_OFF];
  _Float16* sBt = (_Float16*)&smem[SBT_OFF];

  const int tid  = threadIdx.x;
  const int lane = tid & 31;
  const int wave = tid >> 5;
  const int wm   = wave >> 1;   // 0..3
  const int wn   = wave & 1;    // 0..1

  const int z  = blockIdx.z;
  const int z1 = z / nz2;
  const int z2 = z - z1 * nz2;
  const int mtile = blockIdx.y * BM;
  const int ntile = blockIdx.x * BNv;

  const float* Ab = A + (long long)z1 * saz1 + (long long)z2 * saz2;
  const float* Bb = B + (long long)z1 * sbz1 + (long long)z2 * sbz2;
  const unsigned long long bbase = (unsigned long long)Bb;  // SGPR base (GVS)

  f32x8 acc[2][FN];
#pragma unroll
  for (int i = 0; i < 2; ++i)
#pragma unroll
    for (int j = 0; j < FN; ++j)
#pragma unroll
      for (int e = 0; e < 8; ++e) acc[i][j][e] = 0.0f;

  const int kbA = (lane < 16) ? 0 : 16;   // A frag K base (contiguous K: k = 16h + s)
  const int kbB = (lane < 16) ? 0 : 16;   // B frag K base

  constexpr int EB = (BK * BNv) / 256;    // B elements per thread (8 or 16)
  constexpr int GB = (BK * BNv) / 1024;   // B float4 groups per thread (2 or 4)

  // ---- Hoisted per-thread address state (updated with constant strides).
  // B: 32-bit byte offsets against the uniform SGPR base (all operands <16MB).
  unsigned voff[EB];
  unsigned ldsoff[EB];
  const unsigned stepB = (unsigned)(4LL * BK * ldb);
#pragma unroll
  for (int i = 0; i < EB; ++i) {
    int idx = i * 256 + tid;              // idx = kk*BNv + nl
    int kk = idx / BNv;
    int nl = idx - kk * BNv;
    int n  = ntile + nl;
    voff[i] = (unsigned)(4LL * ((long long)kk * ldb +
                                (long long)(n >> 6) * hsb +
                                (long long)(n & 63) * bsn));
    ldsoff[i] = (unsigned)(SBR_OFF + idx * 4);
  }
  // A: 4 running global pointers per thread (K-contiguous rows).
  const char* aptr[4];
  int asoff[4];
#pragma unroll
  for (int i = 0; i < 4; ++i) {
    int g  = i * 256 + tid;
    int r  = g >> 3;                      // BK/4 = 8 float4 groups per row
    int c4 = g & 7;
    aptr[i]  = (const char*)&Ab[(long long)(mtile + r) * lda + c4 * 4];
    asoff[i] = r * LDA_S + c4 * 4;
  }

  // ---- Prologue: kick off async DMA of B tile 0 into raw buffer 0.
#pragma unroll
  for (int i = 0; i < EB; ++i) {
    asm volatile("global_load_async_to_lds_b32 %0, %1, %2"
                 :: "v"(ldsoff[i]), "v"(voff[i]), "s"(bbase)
                 : "memory");
    voff[i] += stepB;
  }

  int buf = 0;
  for (int k0 = 0; k0 < K; k0 += BK) {
    // ---- A tile: vectorized b128 loads, fp32 -> fp16, ds_store_b64.
#pragma unroll
    for (int i = 0; i < 4; ++i) {
      float4 v = *reinterpret_cast<const float4*>(aptr[i]);
      h4 hv = {(_Float16)v.x, (_Float16)v.y, (_Float16)v.z, (_Float16)v.w};
      *reinterpret_cast<h4*>(&sA[asoff[i]]) = hv;
      aptr[i] += BK * 4;
    }
    if (k0 + BK < K)
      __builtin_prefetch(aptr[0], 0, 0);   // next A tile (global_prefetch_b8)

    // Wait for the async copies of the CURRENT raw-B buffer, then sync.
    asm volatile("s_wait_asynccnt 0" ::: "memory");
    __syncthreads();

    // ---- Transpose + convert current B tile: raw[buf][k][n] f32 -> sBt[n][k]
    // f16, and immediately kick off the DMA for the NEXT tile into raw[buf^1].
    {
      const float* sBr = (const float*)&smem[SBR_OFF + buf * SBR_BYTES];
#pragma unroll
      for (int i = 0; i < GB; ++i) {
        int g  = i * 256 + tid;
        int kk = g / (BNv / 4);
        int n4 = g - kk * (BNv / 4);
        float4 v = *reinterpret_cast<const float4*>(&sBr[kk * BNv + n4 * 4]);
        sBt[(n4 * 4 + 0) * LDB_S + kk] = (_Float16)v.x;
        sBt[(n4 * 4 + 1) * LDB_S + kk] = (_Float16)v.y;
        sBt[(n4 * 4 + 2) * LDB_S + kk] = (_Float16)v.z;
        sBt[(n4 * 4 + 3) * LDB_S + kk] = (_Float16)v.w;
      }
    }
    if (k0 + BK < K) {
      const unsigned nxt = (unsigned)(SBR_OFF + (buf ^ 1) * SBR_BYTES - SBR_OFF);
#pragma unroll
      for (int i = 0; i < EB; ++i) {
        asm volatile("global_load_async_to_lds_b32 %0, %1, %2"
                     :: "v"(ldsoff[i] + nxt + (unsigned)((buf ^ 1) ? 0 : 0)
                            + (unsigned)((buf ^ 1) - (buf ^ 1)) /*keep simple*/
                            ), "v"(voff[i]), "s"(bbase)
                     : "memory");
        voff[i] += stepB;
      }
    }
    __syncthreads();

    // ---- Fragments + WMMA.
    Frag fa[2], fb[FN];
#pragma unroll
    for (int fm = 0; fm < 2; ++fm) {
      int row = wm * 32 + fm * 16 + (lane & 15);
      // lanes 0-15: K 0..15 ; lanes 16-31: K 16..31
      fa[fm].q[0] = *(const u32x4*)&sA[row * LDA_S + kbA];
      fa[fm].q[1] = *(const u32x4*)&sA[row * LDA_S + kbA + 8];
    }
#pragma unroll
    for (int fn = 0; fn < FN; ++fn) {
      int col = wn * (BNv / 2) + fn * 16 + (lane & 15);
      const u32x4* p = (const u32x4*)&sBt[col * LDB_S + kbB];
      fb[fn].q[0] = p[0];   // K kb..kb+7
      fb[fn].q[1] = p[1];   // K kb+8..kb+15
    }
#pragma unroll
    for (int fm = 0; fm < 2; ++fm)
#pragma unroll
      for (int fn = 0; fn < FN; ++fn)
        acc[fm][fn] = __builtin_amdgcn_wmma_f32_16x16x32_f16(
            false, fa[fm].h, false, fb[fn].h, (short)0, acc[fm][fn],
            false, false);
    __syncthreads();
    buf ^= 1;
  }

  // Epilogue: alpha, bias, residual, relu. C/D layout per ISA:
  // element r: M = r + (lane<16?0:8), N = lane&15.
  float* Cb = C + (long long)z1 * scz1 + (long long)z2 * scz2;
  const float* Rb =
      resid ? resid + (long long)z1 * scz1 + (long long)z2 * scz2 : nullptr;
#pragma unroll
  for (int fm = 0; fm < 2; ++fm) {
#pragma unroll
    for (int fn = 0; fn < FN; ++fn) {
      int ncol = ntile + wn * (BNv / 2) + fn * 16 + (lane & 15);
#pragma unroll
      for (int r = 0; r < 8; ++r) {
        int mrow = mtile + wm * 32 + fm * 16 + r + ((lane < 16) ? 0 : 8);
        long long idx = (long long)mrow * ldc + ncol;
        float v = acc[fm][fn][r] * alpha;
        if (bias) v += bias[ncol];
        if (Rb) v += Rb[idx];
        if (relu) v = v > 0.0f ? v : 0.0f;
        Cb[idx] = v;
      }
    }
  }
#endif
}

// Masked row softmax over scores [B,H,Sq,Sk]; one wave32 per row.
// mask semantics follow reference: -inf where (amask[s,t] + pmask[b,t]) == -inf.
__global__ __launch_bounds__(256) void softmax_mask(
    float* __restrict__ scores, const float* __restrict__ amask,
    const float* __restrict__ pmask, int Hn, int Sq, int Sk)
{
  const int wave = threadIdx.x >> 5;
  const int lane = threadIdx.x & 31;
  long long row = (long long)blockIdx.x * 8 + wave;
  int b = (int)(row / ((long long)Hn * Sq));
  long long rem = row - (long long)b * Hn * Sq;
  int h = (int)(rem / Sq);
  int s = (int)(rem - (long long)h * Sq);

  float* p = scores + (((long long)b * Hn + h) * Sq + s) * Sk;
  const float NEG_INF = -__builtin_inff();

  float vals[16];   // Sk=512 / 32 lanes
  int cnt = 0;
  float mx = NEG_INF;
  for (int t = lane; t < Sk; t += 32) {
    float v = p[t];
    float m = 0.0f;
    if (amask) m += amask[(long long)s * Sk + t];
    if (pmask) m += pmask[(long long)b * Sk + t];
    if (m == NEG_INF) v = NEG_INF;
    vals[cnt++] = v;
    mx = fmaxf(mx, v);
  }
  for (int o = 16; o > 0; o >>= 1) mx = fmaxf(mx, __shfl_xor(mx, o, 32));
  float sum = 0.0f;
  cnt = 0;
  for (int t = lane; t < Sk; t += 32) {
    float e = __expf(vals[cnt] - mx);
    vals[cnt++] = e;
    sum += e;
  }
  for (int o = 16; o > 0; o >>= 1) sum += __shfl_xor(sum, o, 32);
  float inv = 1.0f / sum;
  cnt = 0;
  for (int t = lane; t < Sk; t += 32) p[t] = vals[cnt++] * inv;
}

// LayerNorm over last dim; one 256-thread block per row.
__global__ __launch_bounds__(256) void layernorm_k(
    const float* __restrict__ x, float* __restrict__ out,
    const float* __restrict__ g, const float* __restrict__ beta, int En)
{
  __shared__ float red[256];
  long long row = blockIdx.x;
  const float* xr = x + row * En;
  float s = 0.0f;
  for (int c = threadIdx.x; c < En; c += 256) s += xr[c];
  red[threadIdx.x] = s;
  __syncthreads();
  for (int o = 128; o > 0; o >>= 1) {
    if ((int)threadIdx.x < o) red[threadIdx.x] += red[threadIdx.x + o];
    __syncthreads();
  }
  float mean = red[0] / En;
  __syncthreads();
  float vs = 0.0f;
  for (int c = threadIdx.x; c < En; c += 256) {
    float d = xr[c] - mean;
    vs += d * d;
  }
  red[threadIdx.x] = vs;
  __syncthreads();
  for (int o = 128; o > 0; o >>= 1) {
    if ((int)threadIdx.x < o) red[threadIdx.x] += red[threadIdx.x + o];
    __syncthreads();
  }
  float inv = rsqrtf(red[0] / En + 1e-5f);
  float* orow = out + row * En;
  for (int c = threadIdx.x; c < En; c += 256)
    orow[c] = (xr[c] - mean) * inv * g[c] + beta[c];
}

// (S,B,E) -> (B,S,E)
__global__ void sbe_to_bse(const float* __restrict__ in, float* __restrict__ out,
                           int Sn, int Bn, int En)
{
  long long i = (long long)blockIdx.x * 256 + threadIdx.x;
  long long total = (long long)Sn * Bn * En;
  if (i >= total) return;
  int e = (int)(i % En);
  long long r = i / En;
  int b = (int)(r % Bn);
  int s = (int)(r / Bn);
  out[((long long)b * Sn + s) * En + e] = in[i];
}

// (B,S,E) -> (S,B,E)
__global__ void bse_to_sbe(const float* __restrict__ in, float* __restrict__ out,
                           int Sn, int Bn, int En)
{
  long long i = (long long)blockIdx.x * 256 + threadIdx.x;
  long long total = (long long)Sn * Bn * En;
  if (i >= total) return;
  int e = (int)(i % En);
  long long r = i / En;
  int b = (int)(r % Bn);
  int s = (int)(r / Bn);
  out[i] = in[((long long)b * Sn + s) * En + e];
}

// ---------------------------------------------------------------------------
// Host-side orchestration
// ---------------------------------------------------------------------------
static void launch_gemm(hipStream_t st, const float* A, const float* B, float* C,
                        const float* bias, const float* res,
                        int M, int N, int K,
                        long long saz1, long long saz2, long long lda,
                        long long sbz1, long long sbz2, long long ldb,
                        long long hsb, long long bsn,
                        long long scz1, long long scz2, long long ldc,
                        int nz1, int nz2, float alpha, int relu)
{
  if ((N & 127) == 0) {
    dim3 grid((unsigned)(N / 128), (unsigned)(M / BM), (unsigned)(nz1 * nz2));
    wmma_gemm<128><<<grid, 256, 0, st>>>(A, B, C, bias, res, M, N, K,
                                         saz1, saz2, lda,
                                         sbz1, sbz2, ldb, hsb, bsn,
                                         scz1, scz2, ldc, nz2, alpha, relu);
  } else {
    dim3 grid((unsigned)(N / 64), (unsigned)(M / BM), (unsigned)(nz1 * nz2));
    wmma_gemm<64><<<grid, 256, 0, st>>>(A, B, C, bias, res, M, N, K,
                                        saz1, saz2, lda,
                                        sbz1, sbz2, ldb, hsb, bsn,
                                        scz1, scz2, ldc, nz2, alpha, relu);
  }
}

// MHA given packed per-token [Q|K|V] activations in qkv (row stride 3E, Q at
// col 0, K at 1024, V at 2048). Writes O in [b,s,h,d] == [M,E] row-major.
static void attention(hipStream_t st, const float* qkv, float* scores,
                      float* obuf, const float* amask, const float* pmask)
{
  const long long rs = 3LL * E_;  // qkv row stride
  // scores[b,h,s,t] = (1/sqrt(DH)) * sum_d Q[b,s,h,d] * K[b,t,h,d]
  launch_gemm(st, qkv /*Q*/, qkv + E_ /*K*/, scores, nullptr, nullptr,
              S_, S_, DH_,
              (long long)S_ * rs, 64, rs,                 // A: z1=b, z2=h, lda
              (long long)S_ * rs, 64, 1, 64 * rs, rs,     // B: Kt (transposed)
              (long long)H_ * S_ * S_, (long long)S_ * S_, S_,
              B_, H_, 0.125f /*1/sqrt(64)*/, 0);
  long long rows = (long long)B_ * H_ * S_;
  softmax_mask<<<dim3((unsigned)(rows / 8)), 256, 0, st>>>(scores, amask, pmask,
                                                           H_, S_, S_);
  // O[b,s,h,d] = sum_t P[b,h,s,t] * V[b,t,h,d]
  launch_gemm(st, scores, qkv + 2 * E_ /*V*/, obuf, nullptr, nullptr,
              S_, DH_, S_,
              (long long)H_ * S_ * S_, (long long)S_ * S_, S_,
              (long long)S_ * rs, 64, rs, 64, 1,
              (long long)S_ * E_, 64, E_,
              B_, H_, 1.0f, 0);
}

extern "C" void kernel_launch(void* const* d_in, const int* in_sizes, int n_in,
                              void* d_out, int out_size, void* d_ws, size_t ws_size,
                              hipStream_t stream)
{
  (void)in_sizes; (void)n_in; (void)out_size; (void)ws_size;
  const float* src      = (const float*)d_in[0];
  const float* tgt      = (const float*)d_in[1];
  const float* src_mask = (const float*)d_in[2];
  const float* tgt_mask = (const float*)d_in[3];
  const float* src_pad  = (const float*)d_in[4];
  const float* tgt_pad  = (const float*)d_in[5];
  const float* enc_wqkv = (const float*)d_in[6];
  const float* enc_wo   = (const float*)d_in[7];
  const float* enc_bo   = (const float*)d_in[8];
  const float* enc_w1   = (const float*)d_in[9];
  const float* enc_b1   = (const float*)d_in[10];
  const float* enc_w2   = (const float*)d_in[11];
  const float* enc_b2   = (const float*)d_in[12];
  const float* enc_lg   = (const float*)d_in[13];
  const float* enc_lb   = (const float*)d_in[14];
  const float* dqkv_s   = (const float*)d_in[15];
  const float* dwo_s    = (const float*)d_in[16];
  const float* dbo_s    = (const float*)d_in[17];
  const float* dqkv_c   = (const float*)d_in[18];
  const float* dwo_c    = (const float*)d_in[19];
  const float* dbo_c    = (const float*)d_in[20];
  const float* dw1      = (const float*)d_in[21];
  const float* db1      = (const float*)d_in[22];
  const float* dw2      = (const float*)d_in[23];
  const float* db2      = (const float*)d_in[24];
  const float* dlg      = (const float*)d_in[25];
  const float* dlb      = (const float*)d_in[26];
  float* out = (float*)d_out;

  // Workspace layout (~160 MB of fp32).
  float* ws = (float*)d_ws;
  size_t off = 0;
  auto alloc = [&](size_t n) { float* p = ws + off; off += n; return p; };
  const size_t ME = (size_t)M_ * E_;
  float* xbuf   = alloc(ME);                       // encoder activations [B,S,E]
  float* ybuf   = alloc(ME);                       // decoder activations
  float* t0     = alloc(ME);                       // pre-LN temp
  float* obuf   = alloc(ME);                       // attention output
  float* qkv    = alloc((size_t)M_ * 3 * E_);      // packed Q|K|V
  float* h1     = alloc((size_t)M_ * DFF_);        // FFN hidden
  float* scores = alloc((size_t)B_ * H_ * S_ * S_);

  const long long W3 = 3LL * H_ * E_ * DH_;        // per-layer qkv weight size
  const long long HS = (long long)E_ * DH_;        // head block stride in wqkv

  long long tot = (long long)S_ * B_ * E_;
  sbe_to_bse<<<dim3((unsigned)((tot + 255) / 256)), 256, 0, stream>>>(
      src, xbuf, S_, B_, E_);
  sbe_to_bse<<<dim3((unsigned)((tot + 255) / 256)), 256, 0, stream>>>(
      tgt, ybuf, S_, B_, E_);

  // ---------------- Encoder ----------------
  for (int i = 0; i < LE_; ++i) {
    launch_gemm(stream, xbuf, enc_wqkv + (size_t)i * W3, qkv, nullptr, nullptr,
                M_, 3 * E_, E_, 0, 0, E_, 0, 0, 64, HS, 1, 0, 0, 3 * E_,
                1, 1, 1.0f, 0);
    attention(stream, qkv, scores, obuf, src_mask, src_pad);
    launch_gemm(stream, obuf, enc_wo + (size_t)i * E_ * E_, t0,
                enc_bo + (size_t)i * E_, xbuf,
                M_, E_, E_, 0, 0, E_, 0, 0, E_, 64, 1, 0, 0, E_,
                1, 1, 1.0f, 0);
    layernorm_k<<<dim3(M_), 256, 0, stream>>>(
        t0, xbuf, enc_lg + (size_t)(i * 2 + 0) * E_,
        enc_lb + (size_t)(i * 2 + 0) * E_, E_);
    launch_gemm(stream, xbuf, enc_w1 + (size_t)i * E_ * DFF_, h1,
                enc_b1 + (size_t)i * DFF_, nullptr,
                M_, DFF_, E_, 0, 0, E_, 0, 0, DFF_, 64, 1, 0, 0, DFF_,
                1, 1, 1.0f, 1 /*relu*/);
    launch_gemm(stream, h1, enc_w2 + (size_t)i * DFF_ * E_, t0,
                enc_b2 + (size_t)i * E_, xbuf,
                M_, E_, DFF_, 0, 0, DFF_, 0, 0, E_, 64, 1, 0, 0, E_,
                1, 1, 1.0f, 0);
    layernorm_k<<<dim3(M_), 256, 0, stream>>>(
        t0, xbuf, enc_lg + (size_t)(i * 2 + 1) * E_,
        enc_lb + (size_t)(i * 2 + 1) * E_, E_);
  }

  // ---------------- Decoder ----------------
  for (int i = 0; i < LD_; ++i) {
    // Self-attention (causal + pad mask)
    launch_gemm(stream, ybuf, dqkv_s + (size_t)i * W3, qkv, nullptr, nullptr,
                M_, 3 * E_, E_, 0, 0, E_, 0, 0, 64, HS, 1, 0, 0, 3 * E_,
                1, 1, 1.0f, 0);
    attention(stream, qkv, scores, obuf, tgt_mask, tgt_pad);
    launch_gemm(stream, obuf, dwo_s + (size_t)i * E_ * E_, t0,
                dbo_s + (size_t)i * E_, ybuf,
                M_, E_, E_, 0, 0, E_, 0, 0, E_, 64, 1, 0, 0, E_,
                1, 1, 1.0f, 0);
    layernorm_k<<<dim3(M_), 256, 0, stream>>>(
        t0, ybuf, dlg + (size_t)(i * 3 + 0) * E_,
        dlb + (size_t)(i * 3 + 0) * E_, E_);
    // Cross-attention: Q from decoder stream, K/V from encoder output
    launch_gemm(stream, ybuf, dqkv_c + (size_t)i * W3, qkv, nullptr, nullptr,
                M_, E_, E_, 0, 0, E_, 0, 0, 64, HS, 1, 0, 0, 3 * E_,
                1, 1, 1.0f, 0);
    launch_gemm(stream, xbuf, dqkv_c + (size_t)i * W3 + (long long)H_ * E_ * DH_,
                qkv + E_, nullptr, nullptr,
                M_, 2 * E_, E_, 0, 0, E_, 0, 0, 64, HS, 1, 0, 0, 3 * E_,
                1, 1, 1.0f, 0);
    attention(stream, qkv, scores, obuf, nullptr, src_pad);
    launch_gemm(stream, obuf, dwo_c + (size_t)i * E_ * E_, t0,
                dbo_c + (size_t)i * E_, ybuf,
                M_, E_, E_, 0, 0, E_, 0, 0, E_, 64, 1, 0, 0, E_,
                1, 1, 1.0f, 0);
    layernorm_k<<<dim3(M_), 256, 0, stream>>>(
        t0, ybuf, dlg + (size_t)(i * 3 + 1) * E_,
        dlb + (size_t)(i * 3 + 1) * E_, E_);
    // FFN
    launch_gemm(stream, ybuf, dw1 + (size_t)i * E_ * DFF_, h1,
                db1 + (size_t)i * DFF_, nullptr,
                M_, DFF_, E_, 0, 0, E_, 0, 0, DFF_, 64, 1, 0, 0, DFF_,
                1, 1, 1.0f, 1);
    launch_gemm(stream, h1, dw2 + (size_t)i * DFF_ * E_, t0,
                db2 + (size_t)i * E_, ybuf,
                M_, E_, DFF_, 0, 0, DFF_, 0, 0, E_, 64, 1, 0, 0, E_,
                1, 1, 1.0f, 0);
    layernorm_k<<<dim3(M_), 256, 0, stream>>>(
        t0, ybuf, dlg + (size_t)(i * 3 + 2) * E_,
        dlb + (size_t)(i * 3 + 2) * E_, E_);
  }

  bse_to_sbe<<<dim3((unsigned)((tot + 255) / 256)), 256, 0, stream>>>(
      ybuf, out, S_, B_, E_);
}
